// ResidualIntegrationNetworkRK4_15710990369031
// MI455X (gfx1250) — matrix operand order repaired and checked by reading, not compiled, not run on hardware
//
#include <hip/hip_runtime.h>

// ---------------------------------------------------------------------------
// ResidualIntegrationNetworkRK4 — persistent single-WGP kernel for MI455X.
//
// 2048 sequential RK4 steps x 4 stages; each stage = MLP 2->512->512->1.
// Layer 2 (128x512 @ 512x512) and layer 3 (128x512 @ 512x1, N padded to 16)
// run on v_wmma_f32_16x16x32_bf16; tanh uses hardware v_tanh_f32.
// Each wave register-blocks a 4x4 grid of C tiles: 16 WMMAs per 8 fragment
// loads, cutting W2 L2 traffic from 8x to 2x redundancy (1MB/stage).
// Activations in LDS (320KB/WGP); W2/W3 pre-swizzled bf16 B-fragments in
// workspace, resident in the 192MB L2.
// ---------------------------------------------------------------------------

typedef __attribute__((ext_vector_type(16))) __bf16        v16bf;
typedef __attribute__((ext_vector_type(8)))  float         v8f;
typedef __attribute__((ext_vector_type(4)))  unsigned int  v4u;

union ABu { v4u u[2]; v16bf v; };      // 32B = one A/B fragment per lane
union Pk  { __bf16 h[2]; unsigned int u; };

#define S_STEPS 2048
#define BATCH   128
#define HDIM    512
#define FDT     1.0f

#define H1_STRIDE 520                   // bf16 elems per row (bank-conflict pad)
#define ROW_BYTES (H1_STRIDE * 2)       // 1040 B, 16B aligned

// ---- dynamic-LDS byte offsets ----
#define OFF_H1   0
#define OFF_H2   (OFF_H1 + BATCH * ROW_BYTES)       // 133120
#define OFF_W1   (OFF_H2 + BATCH * ROW_BYTES)       // 266240
#define OFF_B1   (OFF_W1 + 1024 * 4)
#define OFF_B2   (OFF_B1 + 512 * 4)
#define OFF_XV   (OFF_B2 + 512 * 4)
#define OFF_Y0   (OFF_XV + 128 * 4)
#define OFF_YE   (OFF_Y0 + 128 * 4)
#define OFF_KS   (OFF_YE + 128 * 4)
#define OFF_KC   (OFF_KS + 128 * 4)
#define LDS_BYTES (OFF_KC + 128 * 4)                // 276,992 B < 320KB

// ---- fast tanh: hardware v_tanh_f32 if available, else exp2/rcp approx ----
#if __has_builtin(__builtin_amdgcn_tanhf)
static __device__ __forceinline__ float fast_tanh(float x) {
    return __builtin_amdgcn_tanhf(x);
}
#elif __has_builtin(__builtin_amdgcn_tanh_f32)
static __device__ __forceinline__ float fast_tanh(float x) {
    return __builtin_amdgcn_tanh_f32(x);
}
#else
static __device__ __forceinline__ float fast_tanh(float x) {
    float ax = __builtin_fabsf(x);
    float e  = __builtin_amdgcn_exp2f(ax * -2.8853900817779268f); // -2*log2(e)
    float r  = (1.f - e) * __builtin_amdgcn_rcpf(1.f + e);
    return __builtin_copysignf(r, x);
}
#endif

static __device__ __forceinline__ unsigned short f2bf(float f) {
    union { float f; unsigned int u; } x; x.f = f;
    unsigned int r = x.u + 0x7FFFu + ((x.u >> 16) & 1u);   // RNE
    return (unsigned short)(r >> 16);
}

// ---------------------------------------------------------------------------
// Pre-swizzle W2 (512x512 fp32 [k][n]) into bf16 WMMA B-fragments.
// dword index: ((kt*32 + nt)*32 + lane)*8 + v ; lane: N = nt*16+(lane&15),
// K-base = kt*32 + 8*(lane>>4); dword v covers K = base+2v+(v>=4?8:0), +1.
// ---------------------------------------------------------------------------
__global__ void prep_w2_frag(const float* __restrict__ W2,
                             unsigned int* __restrict__ frag) {
    int idx  = blockIdx.x * 256 + threadIdx.x;       // 0 .. 131071
    int v    = idx & 7;
    int lane = (idx >> 3) & 31;
    int nt   = (idx >> 8) & 31;
    int kt   = idx >> 13;
    int n    = nt * 16 + (lane & 15);
    int k    = kt * 32 + ((lane & 16) >> 1) + 2 * v + ((v & 4) << 1);
    unsigned int lo = f2bf(W2[(size_t)k * HDIM + n]);
    unsigned int hi = f2bf(W2[(size_t)(k + 1) * HDIM + n]);
    frag[idx] = lo | (hi << 16);
}

// W3 (512x1) -> one N-tile of B-fragments, columns 1..15 zero-padded.
__global__ void prep_w3_frag(const float* __restrict__ W3,
                             unsigned int* __restrict__ frag) {
    int idx  = blockIdx.x * 256 + threadIdx.x;       // 0 .. 4095
    int v    = idx & 7;
    int lane = (idx >> 3) & 31;
    int kt   = idx >> 8;
    int n    = lane & 15;
    int k    = kt * 32 + ((lane & 16) >> 1) + 2 * v + ((v & 4) << 1);
    unsigned int lo = (n == 0) ? (unsigned int)f2bf(W3[k])     : 0u;
    unsigned int hi = (n == 0) ? (unsigned int)f2bf(W3[k + 1]) : 0u;
    frag[idx] = lo | (hi << 16);
}

// ---------------------------------------------------------------------------
// Persistent RK4 kernel: 1 block x 256 threads (8 wave32).
// Layer-2 wave map: mtg = wave&1 -> M-tiles [mtg*4, mtg*4+4);
//                   ntg = wave>>1 -> N-tiles [ntg*8, ntg*8+8) in two 4-blocks.
// ---------------------------------------------------------------------------
__global__ void __launch_bounds__(256, 1)
rk4_persistent(const float* __restrict__ x,
               const float* __restrict__ W1, const float* __restrict__ b1,
               const float* __restrict__ b2, const float* __restrict__ b3,
               const v4u*   __restrict__ w2frag,
               const v4u*   __restrict__ w3frag,
               float* __restrict__ out) {
    extern __shared__ char sm[];
    __bf16* h2 = (__bf16*)(sm + OFF_H2);
    float* sW1 = (float*)(sm + OFF_W1);
    float* sb1 = (float*)(sm + OFF_B1);
    float* sb2 = (float*)(sm + OFF_B2);
    float* sxv = (float*)(sm + OFF_XV);
    float* y0  = (float*)(sm + OFF_Y0);
    float* ye  = (float*)(sm + OFF_YE);
    float* ks  = (float*)(sm + OFF_KS);
    float* kc  = (float*)(sm + OFF_KC);

    const int t    = threadIdx.x;
    const int wave = t >> 5;
    const int lane = t & 31;
    const int rl   = lane & 15;       // A: M row in tile / B,C,D: N column
    const int hih  = lane >> 4;       // high half-wave selector
    const int mtg  = wave & 1;        // layer-2 M-tile group
    const int ntg  = wave >> 1;       // layer-2 N-tile group

    for (int i = t; i < 1024; i += 256) sW1[i] = W1[i];
    for (int i = t; i < 512;  i += 256) { sb1[i] = b1[i]; sb2[i] = b2[i]; }
    if (t < 128) { y0[t] = 0.f; ye[t] = 0.f; ks[t] = 0.f; }
    const float b3v = b3[0];
    __syncthreads();

    // Per-im A-row base pointers for layer 2 (M rows mtg*64 + im*16 + rl).
    const char* arowB[4];
#pragma unroll
    for (int im = 0; im < 4; ++im)
        arowB[im] = sm + OFF_H1
                  + (size_t)((mtg * 4 + im) * 16 + rl) * ROW_BYTES + hih * 16;

#pragma unroll 1
    for (int s = 0; s < S_STEPS; ++s) {
        if (t < 128) sxv[t] = x[(size_t)s * BATCH + t];   // v_in, stage-invariant
        __syncthreads();

#pragma unroll 1
        for (int st = 0; st < 4; ++st) {
            // ---- layer 1: h1 = tanh([v,y] @ W1 + b1) -> bf16 LDS ---------
            {
                const int b  = t >> 1;
                const int j0 = (t & 1) * 256;
                const float vin = sxv[b];
                const float yb  = ye[b];
                unsigned int* drow =
                    (unsigned int*)(sm + OFF_H1 + (size_t)b * ROW_BYTES) + (j0 >> 1);
                const float* w1a = sW1 + j0;
                const float* w1b = sW1 + HDIM + j0;
                const float* bb1 = sb1 + j0;
                for (int j = 0; j < 256; j += 2) {
                    float p0 = vin * w1a[j]     + yb * w1b[j]     + bb1[j];
                    float p1 = vin * w1a[j + 1] + yb * w1b[j + 1] + bb1[j + 1];
                    Pk pk;
                    pk.h[0] = (__bf16)fast_tanh(p0);
                    pk.h[1] = (__bf16)fast_tanh(p1);
                    drow[j >> 1] = pk.u;
                }
            }
            __syncthreads();

            // ---- layer 2: h2 = tanh(h1 @ W2 + b2), 4x4 register block ----
#pragma unroll 1
            for (int nb = 0; nb < 2; ++nb) {
                const int ntbase = ntg * 8 + nb * 4;
                v8f acc[16] = {};
                for (int kt = 0; kt < 16; ++kt) {
                    ABu A[4];
#pragma unroll
                    for (int im = 0; im < 4; ++im) {
                        const v4u* ap = (const v4u*)(arowB[im] + kt * 64);
                        A[im].u[0] = ap[0];            // K base .. base+7
                        A[im].u[1] = ap[2];            // K base+16 .. base+23
                    }
#pragma unroll
                    for (int in = 0; in < 4; ++in) {
                        ABu Bm;
                        const v4u* bp = w2frag
                            + ((size_t)(kt * 32 + ntbase + in) * 32 + lane) * 2;
                        Bm.u[0] = bp[0];
                        Bm.u[1] = bp[1];
#pragma unroll
                        for (int im = 0; im < 4; ++im)
                            acc[im * 4 + in] = __builtin_amdgcn_wmma_f32_16x16x32_bf16(
                                false, A[im].v, false, Bm.v, (short)0,
                                acc[im * 4 + in], false, false);
                    }
                }
#pragma unroll
                for (int in = 0; in < 4; ++in) {
                    const int ncol = (ntbase + in) * 16 + rl;
                    const float bias = sb2[ncol];
#pragma unroll
                    for (int im = 0; im < 4; ++im) {
#pragma unroll
                        for (int vv = 0; vv < 8; ++vv) {
                            const int m = (mtg * 4 + im) * 16 + vv + hih * 8;
                            h2[m * H1_STRIDE + ncol] =
                                (__bf16)fast_tanh(acc[im * 4 + in][vv] + bias);
                        }
                    }
                }
            }
            __syncthreads();

            // ---- layer 3: k = h2 @ W3 (+b3) via WMMA, N padded to 16 -----
            {
                const int mt3 = wave;                 // one M-tile per wave
                v8f a3 = {};
                const char* arow2 = sm + OFF_H2
                    + (size_t)(mt3 * 16 + rl) * ROW_BYTES + hih * 16;
                for (int kt = 0; kt < 16; ++kt) {
                    ABu a;
                    const v4u* ap = (const v4u*)(arow2 + kt * 64);
                    a.u[0] = ap[0];
                    a.u[1] = ap[2];
                    ABu bm;
                    const v4u* bp = w3frag + ((size_t)kt * 32 + lane) * 2;
                    bm.u[0] = bp[0];
                    bm.u[1] = bp[1];
                    a3 = __builtin_amdgcn_wmma_f32_16x16x32_bf16(
                        false, a.v, false, bm.v, (short)0, a3, false, false);
                }
                if (rl == 0) {                        // column N==0: lanes 0,16
#pragma unroll
                    for (int vv = 0; vv < 8; ++vv)
                        kc[mt3 * 16 + vv + hih * 8] = a3[vv];
                }
            }
            __syncthreads();

            // ---- RK4 stage bookkeeping ----------------------------------
            if (t < 128) {
                const float kv = kc[t] + b3v;
                const float wgt = (st == 1 || st == 2) ? 2.f : 1.f;
                ks[t] += wgt * kv;
                if (st < 3) {
                    const float cst = (st < 2) ? 0.5f * FDT : FDT;
                    ye[t] = y0[t] + cst * kv;
                }
            }
            __syncthreads();
        }

        if (t < 128) {
            const float y1 = y0[t] + ks[t] * (FDT / 6.f);
            y0[t] = y1; ye[t] = y1; ks[t] = 0.f;
            out[(size_t)s * BATCH + t] = y1;
        }
        __syncthreads();
    }
}

// ---------------------------------------------------------------------------
extern "C" void kernel_launch(void* const* d_in, const int* in_sizes, int n_in,
                              void* d_out, int out_size, void* d_ws, size_t ws_size,
                              hipStream_t stream) {
    (void)in_sizes; (void)n_in; (void)out_size; (void)ws_size;
    const float* x  = (const float*)d_in[0];   // (2048,128,1)
    const float* W1 = (const float*)d_in[1];   // (2,512)
    const float* b1 = (const float*)d_in[2];   // (512)
    const float* W2 = (const float*)d_in[3];   // (512,512)
    const float* b2 = (const float*)d_in[4];   // (512)
    const float* W3 = (const float*)d_in[5];   // (512,1)
    const float* b3 = (const float*)d_in[6];   // (1)
    float* out = (float*)d_out;

    unsigned int* w2f = (unsigned int*)d_ws;            // 512KB B-fragments
    unsigned int* w3f = w2f + (size_t)131072;           // +16KB B-fragments

    prep_w2_frag<<<512, 256, 0, stream>>>(W2, w2f);
    prep_w3_frag<<<16, 256, 0, stream>>>(W3, w3f);
    rk4_persistent<<<1, 256, LDS_BYTES, stream>>>(
        x, W1, b1, b2, b3, (const v4u*)w2f, (const v4u*)w3f, out);
}
